// FeedForwardQuantum_65481071395484
// MI455X (gfx1250) — compile-verified, matrix operand order checked by reading, not run on hardware
//
#include <hip/hip_runtime.h>
#include <hip/hip_bf16.h>

typedef __attribute__((ext_vector_type(16))) _Float16 v16h;
typedef __attribute__((ext_vector_type(8)))  _Float16 half8;
typedef __attribute__((ext_vector_type(4)))  _Float16 half4;
typedef __attribute__((ext_vector_type(8)))  float    v8f;

// Problem constants
static constexpr int EMBED  = 768;
static constexpr int NQ     = 8;
static constexpr int FFN    = 3072;
static constexpr int TOKENS = 8 * 4096;   // B * S = 32768

// Tiling
static constexpr int BM  = 128;   // tokens per block
static constexpr int BN  = 128;   // embed cols per block
static constexpr int BK  = 64;    // FFN k-slab (two WMMA k-steps of 32)
static constexpr int PAD = 8;     // halves of row padding (16B) to spread LDS banks

__global__ __launch_bounds__(256)
void ffq_fused_kernel(const float* __restrict__ x,
                      const float* __restrict__ theta,
                      const float* __restrict__ W1,
                      const float* __restrict__ b1,
                      const float* __restrict__ W2,
                      const float* __restrict__ b2,
                      float* __restrict__ out)
{
    __shared__ float    sQ[BM][NQ];          // 4 KB   q = cos(x)*cos(theta)
    __shared__ _Float16 sA[BM][BK + PAD];    // 18 KB  h-tile (f16, row-major)
    __shared__ _Float16 sB[BN][BK + PAD];    // 18 KB  W2-tile (f16, row = out col)

    const int tid  = threadIdx.x;
    const int tok0 = blockIdx.x * BM;        // first token of this block
    const int n0   = blockIdx.y * BN;        // first output column of this block

    // ---- Stage 0: quantum closed form q[m][j] = cos(x[m][j]) * cos(theta[j]) ----
    for (int i = tid; i < BM * NQ; i += 256) {
        const int m = i >> 3;
        const int j = i & 7;
        sQ[m][j] = __cosf(x[(size_t)(tok0 + m) * EMBED + j]) * __cosf(theta[j]);
    }

    // Wave / lane decomposition: 8 waves as 4 (M) x 2 (N)
    const int lane    = tid & 31;
    const int wave    = tid >> 5;
    const int waveM   = wave >> 1;           // 0..3 -> 32 rows each
    const int waveN   = wave & 1;            // 0..1 -> 64 cols each
    const int rA      = waveM * 32;
    const int cB      = waveN * 64;
    const int r16     = lane & 15;
    const int halfSel = lane >> 4;           // 0: K base, 1: K base + 8

    // Stage-1 work split: 4 consecutive k-columns x 8 rows per thread
    const int kk4   = (tid & 15) * 4;        // 0..60
    const int mrow0 = (tid >> 4) * 8;        // 0..120

    v8f acc[2][4] = {};                      // 8 accumulators (64 VGPRs)

    for (int k0 = 0; k0 < FFN; k0 += BK) {
        // ---- Stage 1: h-tile = relu(q @ W1^T + b1), f32 -> f16 into LDS ----
        {
            float w[4][8];
            float bb[4];
            const int kcol = k0 + kk4;
            #pragma unroll
            for (int c = 0; c < 4; ++c) {
                const float4* wp = (const float4*)&W1[(size_t)(kcol + c) * NQ];
                const float4 wa = wp[0];
                const float4 wz = wp[1];
                w[c][0] = wa.x; w[c][1] = wa.y; w[c][2] = wa.z; w[c][3] = wa.w;
                w[c][4] = wz.x; w[c][5] = wz.y; w[c][6] = wz.z; w[c][7] = wz.w;
                bb[c] = b1[kcol + c];
            }
            #pragma unroll
            for (int r = 0; r < 8; ++r) {
                const int m = mrow0 + r;
                const float4 qa = *(const float4*)&sQ[m][0];
                const float4 qb = *(const float4*)&sQ[m][4];
                const float qv[8] = {qa.x, qa.y, qa.z, qa.w, qb.x, qb.y, qb.z, qb.w};
                half4 hv;
                #pragma unroll
                for (int c = 0; c < 4; ++c) {
                    float a = bb[c];
                    #pragma unroll
                    for (int j = 0; j < NQ; ++j) a = fmaf(qv[j], w[c][j], a);
                    a = fmaxf(a, 0.0f);
                    hv[c] = (_Float16)a;
                }
                *(half4*)&sA[m][kk4] = hv;
            }
        }

        // ---- Stage 2: W2 tile f32 -> f16 into LDS (B: row = output column) ----
        for (int i = tid; i < BN * (BK / 4); i += 256) {
            const int n  = i >> 4;           // 0..127
            const int kq = (i & 15) * 4;     // 0..60
            const float4 f = *(const float4*)&W2[(size_t)(n0 + n) * FFN + k0 + kq];
            half4 hv;
            hv[0] = (_Float16)f.x; hv[1] = (_Float16)f.y;
            hv[2] = (_Float16)f.z; hv[3] = (_Float16)f.w;
            *(half4*)&sB[n][kq] = hv;
        }

        // Prefetch next W2 k-slab into cache while we crunch this one
        if (k0 + BK < FFN) {
            const int n = tid >> 1;
            __builtin_prefetch(&W2[(size_t)(n0 + n) * FFN + (k0 + BK) + (tid & 1) * 32], 0, 0);
        }

        __syncthreads();

        // ---- Stage 3: WMMA over two k-steps of 32 ----
        #pragma unroll
        for (int ks = 0; ks < 2; ++ks) {
            const int kb = ks * 32 + halfSel * 8;   // per-lane K base (ISA 16-bit A layout)
            v16h af[2], bf[4];
            #pragma unroll
            for (int tm = 0; tm < 2; ++tm) {
                const _Float16* p = &sA[rA + tm * 16 + r16][kb];
                const half8 lo = *(const half8*)p;          // K .. K+7
                const half8 hi = *(const half8*)(p + 16);   // K+16 .. K+23
                af[tm] = __builtin_shufflevector(lo, hi,
                         0,1,2,3,4,5,6,7,8,9,10,11,12,13,14,15);
            }
            #pragma unroll
            for (int tn = 0; tn < 4; ++tn) {
                const _Float16* p = &sB[cB + tn * 16 + r16][kb];
                const half8 lo = *(const half8*)p;
                const half8 hi = *(const half8*)(p + 16);
                bf[tn] = __builtin_shufflevector(lo, hi,
                         0,1,2,3,4,5,6,7,8,9,10,11,12,13,14,15);
            }
            #pragma unroll
            for (int tm = 0; tm < 2; ++tm)
                #pragma unroll
                for (int tn = 0; tn < 4; ++tn)
                    acc[tm][tn] = __builtin_amdgcn_wmma_f32_16x16x32_f16(
                        false, af[tm], false, bf[tn],
                        (short)0, acc[tm][tn], false, false);
        }

        __syncthreads();   // protect LDS tiles before next slab overwrites
    }

    // ---- Epilogue: D layout -> global, + b2 ----
    #pragma unroll
    for (int tn = 0; tn < 4; ++tn) {
        const int col  = n0 + cB + tn * 16 + r16;
        const float bias = b2[col];
        #pragma unroll
        for (int tm = 0; tm < 2; ++tm) {
            const int rbase = tok0 + rA + tm * 16 + halfSel * 8;
            #pragma unroll
            for (int v = 0; v < 8; ++v) {
                out[(size_t)(rbase + v) * EMBED + col] = acc[tm][tn][v] + bias;
            }
        }
    }
}

extern "C" void kernel_launch(void* const* d_in, const int* in_sizes, int n_in,
                              void* d_out, int out_size, void* d_ws, size_t ws_size,
                              hipStream_t stream) {
    const float* x     = (const float*)d_in[0];
    const float* theta = (const float*)d_in[1];
    const float* W1    = (const float*)d_in[2];
    const float* b1    = (const float*)d_in[3];
    const float* W2    = (const float*)d_in[4];
    const float* b2    = (const float*)d_in[5];
    float* out = (float*)d_out;

    dim3 grid(TOKENS / BM, EMBED / BN);   // 256 x 6 = 1536 workgroups
    ffq_fused_kernel<<<grid, 256, 0, stream>>>(x, theta, W1, b1, W2, b2, out);
}
